// GraspEvaluatorV2_7456063226448
// MI455X (gfx1250) — compile-verified, matrix operand order checked by reading, not run on hardware
//
#include <hip/hip_runtime.h>
#include <hip/hip_bf16.h>

typedef __attribute__((ext_vector_type(16))) _Float16 v16h;
typedef __attribute__((ext_vector_type(8)))  float    v8f;

namespace {

constexpr int NB   = 64;    // batch
constexpr int NPTS = 1024;  // input points
constexpr float EPS = 1e-5f;

// ------------------------------------------------------------------
// Farthest point sampling: one block per batch element.
// ------------------------------------------------------------------
__global__ void fps_kernel(const float* __restrict__ xyz, int n, int npoint,
                           int* __restrict__ out) {
  __shared__ float dist[1024];
  __shared__ float rmax[256];
  __shared__ int   ridx[256];
  __shared__ float fpt[3];
  __shared__ int   far_s;
  const int b = blockIdx.x, t = threadIdx.x;
  const float* p = xyz + (size_t)b * n * 3;
  for (int i = t; i < n; i += 256) dist[i] = 1e10f;
  if (t == 0) far_s = 0;
  __syncthreads();
  for (int it = 0; it < npoint; ++it) {
    if (t == 0) out[b * npoint + it] = far_s;
    if (t < 3) fpt[t] = p[far_s * 3 + t];
    __syncthreads();
    const float fx = fpt[0], fy = fpt[1], fz = fpt[2];
    float bm = -1.0f; int bi = 0;
    for (int i = t; i < n; i += 256) {
      float dx = p[i*3+0]-fx, dy = p[i*3+1]-fy, dz = p[i*3+2]-fz;
      float nd = fminf(dist[i], dx*dx + dy*dy + dz*dz);
      dist[i] = nd;
      if (nd > bm) { bm = nd; bi = i; }
    }
    rmax[t] = bm; ridx[t] = bi;
    __syncthreads();
    for (int s = 128; s > 0; s >>= 1) {
      if (t < s && rmax[t + s] > rmax[t]) { rmax[t] = rmax[t + s]; ridx[t] = ridx[t + s]; }
      __syncthreads();
    }
    if (t == 0) far_s = ridx[0];
    __syncthreads();
  }
}

// gather centroids: thread per (b,s)
__global__ void gather_xyz(const float* __restrict__ xyz, const int* __restrict__ fidx,
                           int n, int S, float* __restrict__ nxyz) {
  int i = blockIdx.x * blockDim.x + threadIdx.x;
  if (i >= NB * S) return;
  int b = i / S;
  int j = fidx[i];
  const float* p = xyz + ((size_t)b * n + j) * 3;
  nxyz[i*3+0] = p[0]; nxyz[i*3+1] = p[1]; nxyz[i*3+2] = p[2];
}

// ball query matching the reference "sorted ascending, pad with first" semantics
__global__ void ball_query(const float* __restrict__ xyz, const float* __restrict__ nxyz,
                           int n, int S, int nsample, float r2, int* __restrict__ gidx) {
  int i = blockIdx.x * blockDim.x + threadIdx.x;
  if (i >= NB * S) return;
  int b = i / S;
  const float* p = xyz + (size_t)b * n * 3;
  float cx = nxyz[i*3+0], cy = nxyz[i*3+1], cz = nxyz[i*3+2];
  int* out = gidx + (size_t)i * nsample;
  int cnt = 0, first = -1;
  for (int j = 0; j < n && cnt < nsample; ++j) {
    float dx = p[j*3+0]-cx, dy = p[j*3+1]-cy, dz = p[j*3+2]-cz;
    if (dx*dx + dy*dy + dz*dz <= r2) {
      if (first < 0) first = j;
      out[cnt++] = j;
    }
  }
  if (first < 0) first = 0;
  for (; cnt < nsample; ++cnt) out[cnt] = first;
}

// ------------------------------------------------------------------
// Grouping kernels: produce f16 activation rows [M, Kpad]
// ------------------------------------------------------------------
__global__ void group_sa1(const float* __restrict__ xyz, const float* __restrict__ nxyz,
                          const int* __restrict__ gidx, _Float16* __restrict__ X) {
  const int S = 128, K = 64, KP = 32;
  int i = blockIdx.x * blockDim.x + threadIdx.x;
  if (i >= NB * S * K) return;
  int b = i / (S * K);
  int s = (i / K) % S;
  int j = gidx[i];
  const float* p = xyz + ((size_t)b * NPTS + j) * 3;
  const float* c = nxyz + ((size_t)b * S + s) * 3;
  _Float16* o = X + (size_t)i * KP;
  o[0] = (_Float16)(p[0] - c[0]);
  o[1] = (_Float16)(p[1] - c[1]);
  o[2] = (_Float16)(p[2] - c[2]);
  for (int q = 3; q < KP; ++q) o[q] = (_Float16)0.f;
}

__global__ void group_sa2(const float* __restrict__ xyz1, const float* __restrict__ feats1,
                          const float* __restrict__ nxyz2, const int* __restrict__ gidx,
                          _Float16* __restrict__ X) {
  const int S = 32, K = 128, KP = 160, N1 = 128, C1 = 128;
  int i = blockIdx.x * blockDim.x + threadIdx.x;
  if (i >= NB * S * K) return;
  int b = i / (S * K);
  int s = (i / K) % S;
  int j = gidx[i];
  const float* p = xyz1 + ((size_t)b * N1 + j) * 3;
  const float* c = nxyz2 + ((size_t)b * S + s) * 3;
  const float* f = feats1 + ((size_t)b * N1 + j) * C1;
  _Float16* o = X + (size_t)i * KP;
  o[0] = (_Float16)(p[0] - c[0]);
  o[1] = (_Float16)(p[1] - c[1]);
  o[2] = (_Float16)(p[2] - c[2]);
  for (int q = 0; q < C1; ++q) o[3 + q] = (_Float16)f[q];
  for (int q = 3 + C1; q < KP; ++q) o[q] = (_Float16)0.f;
}

__global__ void group_sa3(const float* __restrict__ xyz2, const float* __restrict__ feats2,
                          _Float16* __restrict__ X) {
  const int K = 32, KP = 288, C2 = 256;
  int i = blockIdx.x * blockDim.x + threadIdx.x;
  if (i >= NB * K) return;
  const float* p = xyz2 + (size_t)i * 3;
  const float* f = feats2 + (size_t)i * C2;
  _Float16* o = X + (size_t)i * KP;
  o[0] = (_Float16)p[0]; o[1] = (_Float16)p[1]; o[2] = (_Float16)p[2];
  for (int q = 0; q < C2; ++q) o[3 + q] = (_Float16)f[q];
  for (int q = 3 + C2; q < KP; ++q) o[q] = (_Float16)0.f;
}

// weights f32[N][Cin] -> f16[N][Kpad], zero padded
__global__ void wprep(const float* __restrict__ W, int Cin, int Kpad, int N,
                      _Float16* __restrict__ WH) {
  int i = blockIdx.x * blockDim.x + threadIdx.x;
  if (i >= N * Kpad) return;
  int o = i / Kpad, c = i % Kpad;
  WH[i] = (c < Cin) ? (_Float16)W[o * Cin + c] : (_Float16)0.f;
}

__global__ void zero_stats(float* a, float* b2, int N) {
  int i = blockIdx.x * blockDim.x + threadIdx.x;
  if (i < N) { a[i] = 0.f; b2[i] = 0.f; }
}

// ------------------------------------------------------------------
// WMMA fragment loader (16-bit 16x32 A / B-transposed layout, §7.12.2)
// ------------------------------------------------------------------
__device__ __forceinline__ v16h load_frag16x32(const _Float16* __restrict__ base,
                                               int stride, int lane, int k0) {
  const int r  = lane & 15;
  const int kb = k0 + ((lane >> 4) << 3);  // lanes 0-15: +0 ; 16-31: +8
  const _Float16* p = base + (size_t)r * stride + kb;
  v16h v;
#pragma unroll
  for (int i = 0; i < 8; ++i) v[i] = p[i];          // K = kb .. kb+7
#pragma unroll
  for (int i = 0; i < 8; ++i) v[8 + i] = p[16 + i]; // K = kb+16 .. kb+23
  return v;
}

// Each wave owns one 16-row M tile and FOUR consecutive 16-col N tiles:
// the A fragment is loaded once per k-step and reused across 4 WMMAs,
// cutting activation HBM traffic 4x. All layer widths are multiples of 64.

// Pass 1: WMMA GEMM tiles -> per-channel sum / sumsq (for BN stats).
__global__ void gemm_stats(const _Float16* __restrict__ X, const _Float16* __restrict__ WH,
                           int M, int N, int Kpad,
                           float* __restrict__ ssum, float* __restrict__ ssq) {
  const int mt = M >> 4;
  const int wid = blockIdx.x * 8 + (threadIdx.x >> 5);
  if (wid >= mt * (N >> 6)) return;           // wave-uniform exit (EXEC all-1 in WMMA)
  const int lane = threadIdx.x & 31;
  const int m0 = (wid % mt) << 4;
  const int n0 = (wid / mt) << 6;
  const _Float16* Abase = X  + (size_t)m0 * Kpad;
  const _Float16* Bbase = WH + (size_t)n0 * Kpad;
  const v8f z = {0.f,0.f,0.f,0.f,0.f,0.f,0.f,0.f};
  v8f acc[4] = {z, z, z, z};
  for (int k0 = 0; k0 < Kpad; k0 += 32) {
    v16h af = load_frag16x32(Abase, Kpad, lane, k0);
#pragma unroll
    for (int t = 0; t < 4; ++t) {
      v16h bf = load_frag16x32(Bbase + ((size_t)t << 4) * Kpad, Kpad, lane, k0);
      acc[t] = __builtin_amdgcn_wmma_f32_16x16x32_f16(false, af, false, bf,
                                                      (short)0, acc[t], false, false);
    }
  }
#pragma unroll
  for (int t = 0; t < 4; ++t) {
    float s1 = 0.f, s2 = 0.f;
#pragma unroll
    for (int j = 0; j < 8; ++j) { float y = acc[t][j]; s1 += y; s2 += y * y; }
    // lane n and lane n+16 hold the same output channel: combine, halve the atomics
    s1 += __shfl_xor(s1, 16, 32);
    s2 += __shfl_xor(s2, 16, 32);
    if (lane < 16) {
      atomicAdd(&ssum[n0 + (t << 4) + lane], s1);
      atomicAdd(&ssq[n0 + (t << 4) + lane], s2);
    }
  }
}

__global__ void bn_prep(const float* __restrict__ ssum, const float* __restrict__ ssq,
                        const float* __restrict__ g, const float* __restrict__ bt,
                        int N, float count, float* __restrict__ scl, float* __restrict__ shf) {
  int n = blockIdx.x * blockDim.x + threadIdx.x;
  if (n >= N) return;
  float mean = ssum[n] / count;
  float var  = ssq[n] / count - mean * mean;
  float sc   = g[n] * rsqrtf(var + EPS);
  scl[n] = sc;
  shf[n] = bt[n] - mean * sc;
}

// Pass 2: recompute WMMA tiles, apply BN scale/shift + ReLU, emit f16 rows [M, N].
__global__ void gemm_bn_relu(const _Float16* __restrict__ X, const _Float16* __restrict__ WH,
                             int M, int N, int Kpad,
                             const float* __restrict__ scl, const float* __restrict__ shf,
                             _Float16* __restrict__ Xout) {
  const int mt = M >> 4;
  const int wid = blockIdx.x * 8 + (threadIdx.x >> 5);
  if (wid >= mt * (N >> 6)) return;
  const int lane = threadIdx.x & 31;
  const int m0 = (wid % mt) << 4;
  const int n0 = (wid / mt) << 6;
  const _Float16* Abase = X  + (size_t)m0 * Kpad;
  const _Float16* Bbase = WH + (size_t)n0 * Kpad;
  const v8f z = {0.f,0.f,0.f,0.f,0.f,0.f,0.f,0.f};
  v8f acc[4] = {z, z, z, z};
  for (int k0 = 0; k0 < Kpad; k0 += 32) {
    v16h af = load_frag16x32(Abase, Kpad, lane, k0);
#pragma unroll
    for (int t = 0; t < 4; ++t) {
      v16h bf = load_frag16x32(Bbase + ((size_t)t << 4) * Kpad, Kpad, lane, k0);
      acc[t] = __builtin_amdgcn_wmma_f32_16x16x32_f16(false, af, false, bf,
                                                      (short)0, acc[t], false, false);
    }
  }
  const int mh = (lane >> 4) << 3;          // lanes>=16 own rows m0+8..15
#pragma unroll
  for (int t = 0; t < 4; ++t) {
    const int n = n0 + (t << 4) + (lane & 15);
    const float sc = scl[n], sh = shf[n];
#pragma unroll
    for (int j = 0; j < 8; ++j) {
      float y = fmaxf(acc[t][j] * sc + sh, 0.f);
      Xout[(size_t)(m0 + mh + j) * N + n] = (_Float16)y;
    }
  }
}

// max over the group dim k: thread per (b,s,n)
__global__ void maxpool_k(const _Float16* __restrict__ X, int S, int Kg, int N,
                          float* __restrict__ feats) {
  int i = blockIdx.x * blockDim.x + threadIdx.x;
  if (i >= NB * S * N) return;
  int n = i % N;
  int bs = i / N;
  const _Float16* p = X + ((size_t)bs * Kg) * N + n;
  float m = -1e30f;
  for (int k = 0; k < Kg; ++k) m = fmaxf(m, (float)p[(size_t)k * N]);
  feats[i] = m;
}

// ------------------------------------------------------------------
// FC head (tiny)
// ------------------------------------------------------------------
__global__ void concat_head(const float* __restrict__ f3, const float* __restrict__ A,
                            const float* __restrict__ T, float* __restrict__ x) {
  int i = blockIdx.x * blockDim.x + threadIdx.x;
  if (i >= NB * 525) return;
  int r = i / 525, c = i % 525;
  float v;
  if (c < 512)      v = f3[r * 512 + c];
  else if (c < 522) v = A[r * 10 + (c - 512)];
  else              v = T[r * 3 + (c - 522)];
  x[i] = v;
}

__global__ void fc_gemm(const float* __restrict__ x, const float* __restrict__ W,
                        const float* __restrict__ bias, int Cin, int Cout,
                        float* __restrict__ y) {
  int i = blockIdx.x * blockDim.x + threadIdx.x;
  if (i >= NB * Cout) return;
  int r = i / Cout, o = i % Cout;
  const float* xr = x + (size_t)r * Cin;
  const float* wr = W + (size_t)o * Cin;
  float acc = bias[o];
  for (int c = 0; c < Cin; ++c) acc += xr[c] * wr[c];
  y[i] = acc;
}

// BN over batch axis + ReLU: one block per channel, 64 threads (rows)
__global__ void bn_fc(const float* __restrict__ y, const float* __restrict__ g,
                      const float* __restrict__ bt, int Cout, float* __restrict__ xo) {
  __shared__ float s1[64], s2[64];
  int o = blockIdx.x, t = threadIdx.x;
  float v = y[(size_t)t * Cout + o];
  s1[t] = v; s2[t] = v * v;
  __syncthreads();
  for (int s = 32; s > 0; s >>= 1) {
    if (t < s) { s1[t] += s1[t + s]; s2[t] += s2[t + s]; }
    __syncthreads();
  }
  float mean = s1[0] / 64.f;
  float var  = s2[0] / 64.f - mean * mean;
  xo[(size_t)t * Cout + o] = fmaxf(g[o] * (v - mean) * rsqrtf(var + EPS) + bt[o], 0.f);
}

__global__ void cls_gemm(const float* __restrict__ x, const float* __restrict__ W,
                         const float* __restrict__ b, float* __restrict__ out) {
  int r = threadIdx.x;
  if (r >= NB) return;
  const float* xr = x + (size_t)r * 64;
  float acc = b[0];
  for (int c = 0; c < 64; ++c) acc += xr[c] * W[c];
  out[r] = acc;
}

} // namespace

// ------------------------------------------------------------------
extern "C" void kernel_launch(void* const* d_in, const int* in_sizes, int n_in,
                              void* d_out, int out_size, void* d_ws, size_t ws_size,
                              hipStream_t stream) {
  (void)in_sizes; (void)n_in; (void)out_size; (void)ws_size;

  // ---- input mapping (pytree flatten: top dict order, params dict sorted keys) ----
  const float* A_vec = (const float*)d_in[0];
  const float* trans = (const float*)d_in[1];
  const float* pc    = (const float*)d_in[2];
  const float* bn1g = (const float*)d_in[3],  *bn1b = (const float*)d_in[4];
  const float* bn2g = (const float*)d_in[5],  *bn2b = (const float*)d_in[6];
  const float* bn3g = (const float*)d_in[7],  *bn3b = (const float*)d_in[8];
  const float* clsw = (const float*)d_in[9],  *clsb = (const float*)d_in[10];
  const float* fc1w = (const float*)d_in[11], *fc1b = (const float*)d_in[12];
  const float* fc2w = (const float*)d_in[13], *fc2b = (const float*)d_in[14];
  const float* fc3w = (const float*)d_in[15], *fc3b = (const float*)d_in[16];
  // sa[l][layer] = (w, b, g, bt) starting at index 17
  auto saw = [&](int l, int j, int which) -> const float* {
    return (const float*)d_in[17 + (l * 3 + j) * 4 + which];
  };

  // ---- workspace bump allocator ----
  char* wp = (char*)d_ws;
  auto alloc = [&](size_t bytes) -> void* {
    void* r = (void*)wp;
    wp += (bytes + 255) & ~(size_t)255;
    return r;
  };

  const int S1 = 128, K1 = 64;   // SA1
  const int S2 = 32,  K2 = 128;  // SA2
  const int M1 = NB * S1 * K1;   // 524288
  const int M2 = NB * S2 * K2;   // 262144
  const int M3 = NB * 32;        // 2048

  _Float16* Xa = (_Float16*)alloc((size_t)M1 * 128 * sizeof(_Float16));
  _Float16* Xb = (_Float16*)alloc((size_t)M1 * 128 * sizeof(_Float16));
  _Float16* WH = (_Float16*)alloc((size_t)512 * 288 * sizeof(_Float16));
  float* ssum = (float*)alloc(512 * 4);
  float* ssq  = (float*)alloc(512 * 4);
  float* scl  = (float*)alloc(512 * 4);
  float* shf  = (float*)alloc(512 * 4);
  int* fidx1 = (int*)alloc((size_t)NB * S1 * 4);
  int* fidx2 = (int*)alloc((size_t)NB * S2 * 4);
  float* nxyz1 = (float*)alloc((size_t)NB * S1 * 3 * 4);
  float* nxyz2 = (float*)alloc((size_t)NB * S2 * 3 * 4);
  int* gidx1 = (int*)alloc((size_t)NB * S1 * K1 * 4);
  int* gidx2 = (int*)alloc((size_t)NB * S2 * K2 * 4);
  float* feats1 = (float*)alloc((size_t)NB * S1 * 128 * 4);
  float* feats2 = (float*)alloc((size_t)NB * S2 * 256 * 4);
  float* feats3 = (float*)alloc((size_t)NB * 512 * 4);
  float* xh0 = (float*)alloc((size_t)NB * 525 * 4);
  float* hy1 = (float*)alloc((size_t)NB * 256 * 4);
  float* hx1 = (float*)alloc((size_t)NB * 256 * 4);
  float* hy2 = (float*)alloc((size_t)NB * 128 * 4);
  float* hx2 = (float*)alloc((size_t)NB * 128 * 4);
  float* hy3 = (float*)alloc((size_t)NB * 64 * 4);
  float* hx3 = (float*)alloc((size_t)NB * 64 * 4);

  auto cdiv = [](int a, int b) { return (a + b - 1) / b; };

  // one MLP layer: weight prep -> stats pass (WMMA) -> bn coeffs -> apply pass (WMMA)
  auto sa_layer = [&](const _Float16* Xin, _Float16* Xout, int M, int Kpad, int Cin,
                      int Nout, const float* W, const float* g, const float* bt) {
    int tiles = (M >> 4) * (Nout >> 6);   // one wave = 16 rows x 64 cols
    int blocks = cdiv(tiles, 8);
    wprep<<<cdiv(Nout * Kpad, 256), 256, 0, stream>>>(W, Cin, Kpad, Nout, WH);
    zero_stats<<<cdiv(Nout, 256), 256, 0, stream>>>(ssum, ssq, Nout);
    gemm_stats<<<blocks, 256, 0, stream>>>(Xin, WH, M, Nout, Kpad, ssum, ssq);
    bn_prep<<<cdiv(Nout, 64), 64, 0, stream>>>(ssum, ssq, g, bt, Nout, (float)M, scl, shf);
    gemm_bn_relu<<<blocks, 256, 0, stream>>>(Xin, WH, M, Nout, Kpad, scl, shf, Xout);
  };

  // ================= SA1 =================
  fps_kernel<<<NB, 256, 0, stream>>>(pc, NPTS, S1, fidx1);
  gather_xyz<<<cdiv(NB * S1, 256), 256, 0, stream>>>(pc, fidx1, NPTS, S1, nxyz1);
  ball_query<<<cdiv(NB * S1, 256), 256, 0, stream>>>(pc, nxyz1, NPTS, S1, K1,
                                                     0.02f * 0.02f, gidx1);
  group_sa1<<<cdiv(M1, 256), 256, 0, stream>>>(pc, nxyz1, gidx1, Xa);
  sa_layer(Xa, Xb, M1, 32, 3,  64,  saw(0,0,0), saw(0,0,2), saw(0,0,3));
  sa_layer(Xb, Xa, M1, 64, 64, 64,  saw(0,1,0), saw(0,1,2), saw(0,1,3));
  sa_layer(Xa, Xb, M1, 64, 64, 128, saw(0,2,0), saw(0,2,2), saw(0,2,3));
  maxpool_k<<<cdiv(NB * S1 * 128, 256), 256, 0, stream>>>(Xb, S1, K1, 128, feats1);

  // ================= SA2 =================
  fps_kernel<<<NB, 256, 0, stream>>>(nxyz1, S1, S2, fidx2);
  gather_xyz<<<cdiv(NB * S2, 256), 256, 0, stream>>>(nxyz1, fidx2, S1, S2, nxyz2);
  ball_query<<<cdiv(NB * S2, 256), 256, 0, stream>>>(nxyz1, nxyz2, S1, S2, K2,
                                                     0.04f * 0.04f, gidx2);
  group_sa2<<<cdiv(M2, 256), 256, 0, stream>>>(nxyz1, feats1, nxyz2, gidx2, Xa);
  sa_layer(Xa, Xb, M2, 160, 131, 128, saw(1,0,0), saw(1,0,2), saw(1,0,3));
  sa_layer(Xb, Xa, M2, 128, 128, 128, saw(1,1,0), saw(1,1,2), saw(1,1,3));
  sa_layer(Xa, Xb, M2, 128, 128, 256, saw(1,2,0), saw(1,2,2), saw(1,2,3));
  maxpool_k<<<cdiv(NB * S2 * 256, 256), 256, 0, stream>>>(Xb, S2, K2, 256, feats2);

  // ================= SA3 (group_all) =================
  group_sa3<<<cdiv(M3, 256), 256, 0, stream>>>(nxyz2, feats2, Xa);
  sa_layer(Xa, Xb, M3, 288, 259, 256, saw(2,0,0), saw(2,0,2), saw(2,0,3));
  sa_layer(Xb, Xa, M3, 256, 256, 256, saw(2,1,0), saw(2,1,2), saw(2,1,3));
  sa_layer(Xa, Xb, M3, 256, 256, 512, saw(2,2,0), saw(2,2,2), saw(2,2,3));
  maxpool_k<<<cdiv(NB * 512, 256), 256, 0, stream>>>(Xb, 1, 32, 512, feats3);

  // ================= FC head =================
  concat_head<<<cdiv(NB * 525, 256), 256, 0, stream>>>(feats3, A_vec, trans, xh0);
  fc_gemm<<<cdiv(NB * 256, 256), 256, 0, stream>>>(xh0, fc1w, fc1b, 525, 256, hy1);
  bn_fc<<<256, 64, 0, stream>>>(hy1, bn1g, bn1b, 256, hx1);
  fc_gemm<<<cdiv(NB * 128, 256), 256, 0, stream>>>(hx1, fc2w, fc2b, 256, 128, hy2);
  bn_fc<<<128, 64, 0, stream>>>(hy2, bn2g, bn2b, 128, hx2);
  fc_gemm<<<cdiv(NB * 64, 256), 256, 0, stream>>>(hx2, fc3w, fc3b, 128, 64, hy3);
  bn_fc<<<64, 64, 0, stream>>>(hy3, bn3g, bn3b, 64, hx3);
  cls_gemm<<<1, 64, 0, stream>>>(hx3, clsw, clsb, (float*)d_out);
}
